// SimVQ1D_24541443129907
// MI455X (gfx1250) — matrix operand order repaired and checked by reading, not compiled, not run on hardware
//
#include <hip/hip_runtime.h>
#include <stdint.h>

#define N_E   8192
#define EDIM  512
#define TQ    1024
#define MROWS 8192   // B*T
#define LSTR  40     // LDS row stride in bf16 elements (80B, 16B-aligned, bank-conflict-free frags)
#define NBLKS 16     // N split into 16 chunks of 512 codes (aligned with 2048-wide domains)

typedef __attribute__((ext_vector_type(16))) __bf16 v16bf;
typedef __attribute__((ext_vector_type(8)))  __bf16 v8bf;
typedef __attribute__((ext_vector_type(8)))  float  v8f;
typedef __attribute__((ext_vector_type(4)))  int    v4i;

#if __has_builtin(__builtin_amdgcn_global_load_async_to_lds_b128)
#define ASYNC_LDS 1
#else
#define ASYNC_LDS 0
#endif

#define AS1 __attribute__((address_space(1)))
#define AS3 __attribute__((address_space(3)))

// 16-byte global -> LDS copy: async DMA (ASYNCcnt) when available, else via VGPRs
__device__ __forceinline__ void cp16(unsigned short* l, const unsigned short* g) {
#if ASYNC_LDS
  __builtin_amdgcn_global_load_async_to_lds_b128(
      (AS1 v4i*)(uintptr_t)g, (AS3 v4i*)(unsigned)(uintptr_t)l, 0, 0);
#else
  *(uint4*)l = *(const uint4*)g;
#endif
}
__device__ __forceinline__ void wait_async0() {
#if ASYNC_LDS
#if __has_builtin(__builtin_amdgcn_s_wait_asynccnt)
  __builtin_amdgcn_s_wait_asynccnt(0);
#else
  asm volatile("s_wait_asynccnt 0x0" ::: "memory");
#endif
#endif
}

// ---------- bf16 split helpers (round-to-nearest-even) ----------
__device__ __forceinline__ unsigned short f2bf_rn(float x) {
  unsigned u = __float_as_uint(x);
  u += 0x7FFFu + ((u >> 16) & 1u);
  return (unsigned short)(u >> 16);
}
__device__ __forceinline__ float bf2f(unsigned short h) {
  return __uint_as_float(((unsigned)h) << 16);
}
__device__ __forceinline__ void split2(float x, unsigned short& h, unsigned short& l) {
  h = f2bf_rn(x);
  l = f2bf_rn(x - bf2f(h));
}

// ---------- WMMA fragment gathers from LDS (row-major [row][k] tiles) ----------
// A (16x32 bf16): lane L -> row M = L&15; K = {hi8..hi8+7, 16+hi8..16+hi8+7}, hi8=(L>>4)*8
__device__ __forceinline__ v16bf frag_a(const unsigned short* p) {
  v8bf c0 = *(const v8bf*)(p);
  v8bf c1 = *(const v8bf*)(p + 16);
  return __builtin_shufflevector(c0, c1, 0,1,2,3,4,5,6,7,8,9,10,11,12,13,14,15);
}
// B (32x16 bf16): lane L -> col N = L&15; K = {hi8..hi8+7, 16+hi8..16+hi8+7}, hi8=(L>>4)*8
__device__ __forceinline__ v16bf frag_b(const unsigned short* p) {
  v8bf c0 = *(const v8bf*)(p);
  v8bf c1 = *(const v8bf*)(p + 16);
  return __builtin_shufflevector(c0, c1, 0,1,2,3,4,5,6,7,8,9,10,11,12,13,14,15);
}
__device__ __forceinline__ v8f wmma_acc(v16bf a, v16bf b, v8f c) {
  return __builtin_amdgcn_wmma_f32_16x16x32_bf16(false, a, false, b, (short)0, c, false, false);
}

// ---------- Kernel 1: transpose z (B,C,T) -> zf (B*T, C) with hi/lo bf16 split ----------
__global__ __launch_bounds__(256) void k_transpose_split(
    const float* __restrict__ z,
    unsigned short* __restrict__ zf_hi, unsigned short* __restrict__ zf_lo) {
  __shared__ float tile[32][33];
  int tx = threadIdx.x, ty = threadIdx.y;
  int t0 = blockIdx.x * 32, c0 = blockIdx.y * 32, b = blockIdx.z;
  const float* src = z + (size_t)b * EDIM * TQ;
#pragma unroll
  for (int j = 0; j < 4; ++j) {
    int c = c0 + ty + 8 * j;
    tile[ty + 8 * j][tx] = src[(size_t)c * TQ + t0 + tx];
  }
  __syncthreads();
#pragma unroll
  for (int j = 0; j < 4; ++j) {
    int t = t0 + ty + 8 * j;
    int c = c0 + tx;
    float v = tile[tx][ty + 8 * j];
    unsigned short h, l; split2(v, h, l);
    size_t o = (size_t)(b * TQ + t) * EDIM + c;
    zf_hi[o] = h; zf_lo[o] = l;
  }
}

// ---------- Kernel 2: codebook = emb @ proj_w^T + proj_b, hi/lo bf16 out ----------
// Software-pipelined: next K-step's f32 data staged in VGPRs during WMMA compute.
__global__ __launch_bounds__(256) void k_codebook(
    const float* __restrict__ emb, const float* __restrict__ pw, const float* __restrict__ pb,
    unsigned short* __restrict__ cb_hi, unsigned short* __restrict__ cb_lo) {
  __shared__ unsigned short la_h[128 * LSTR], la_l[128 * LSTR];
  __shared__ unsigned short lb_h[128 * LSTR], lb_l[128 * LSTR];
  int tid = threadIdx.x, lane = tid & 31, wave = tid >> 5;
  int wm = wave & 3, wn = wave >> 2;          // 4 (M) x 2 (N) wave grid, wave tile 32x64
  int mBase = blockIdx.x * 128, nBase = blockIdx.y * 128;
  v8f zero8 = {0.f,0.f,0.f,0.f,0.f,0.f,0.f,0.f};
  v8f acc[2][4];
#pragma unroll
  for (int mt = 0; mt < 2; ++mt)
#pragma unroll
    for (int nt = 0; nt < 4; ++nt) acc[mt][nt] = zero8;

  int hi8 = (lane >> 4) * 8, kh16 = (lane >> 4) * 16, ln = lane & 15;

  float4 rA[4], rB[4];
  auto loadg = [&](int ks) {
    int k0 = ks * 32;
#pragma unroll
    for (int i = 0; i < 4; ++i) {
      int idx = tid + 256 * i, row = idx >> 3, q = idx & 7;
      rA[i] = *(const float4*)(emb + (size_t)(mBase + row) * EDIM + k0 + q * 4);
      rB[i] = *(const float4*)(pw  + (size_t)(nBase + row) * EDIM + k0 + q * 4);
    }
  };
  auto store_split = [&]() {
#pragma unroll
    for (int i = 0; i < 4; ++i) {
      int idx = tid + 256 * i, row = idx >> 3, q = idx & 7;
      unsigned short h0,h1,h2,h3,l0,l1,l2,l3;
      split2(rA[i].x,h0,l0); split2(rA[i].y,h1,l1); split2(rA[i].z,h2,l2); split2(rA[i].w,h3,l3);
      *(uint2*)(la_h + row * LSTR + q * 4) =
          make_uint2((unsigned)h0 | ((unsigned)h1 << 16), (unsigned)h2 | ((unsigned)h3 << 16));
      *(uint2*)(la_l + row * LSTR + q * 4) =
          make_uint2((unsigned)l0 | ((unsigned)l1 << 16), (unsigned)l2 | ((unsigned)l3 << 16));
      split2(rB[i].x,h0,l0); split2(rB[i].y,h1,l1); split2(rB[i].z,h2,l2); split2(rB[i].w,h3,l3);
      *(uint2*)(lb_h + row * LSTR + q * 4) =
          make_uint2((unsigned)h0 | ((unsigned)h1 << 16), (unsigned)h2 | ((unsigned)h3 << 16));
      *(uint2*)(lb_l + row * LSTR + q * 4) =
          make_uint2((unsigned)l0 | ((unsigned)l1 << 16), (unsigned)l2 | ((unsigned)l3 << 16));
    }
  };

  loadg(0);
  for (int ks = 0; ks < 16; ++ks) {
    store_split();
    __syncthreads();
    if (ks < 15) loadg(ks + 1);               // overlaps with WMMA below
    v16bf ah[2], al[2];
#pragma unroll
    for (int mt = 0; mt < 2; ++mt) {
      int r = (wm * 32 + mt * 16 + ln) * LSTR + hi8;
      ah[mt] = frag_a(la_h + r);
      al[mt] = frag_a(la_l + r);
    }
#pragma unroll
    for (int nt = 0; nt < 4; ++nt) {
      int r = (wn * 64 + nt * 16 + ln) * LSTR + hi8;
      v16bf bh = frag_b(lb_h + r);
      v16bf bl = frag_b(lb_l + r);
#pragma unroll
      for (int mt = 0; mt < 2; ++mt) {
        acc[mt][nt] = wmma_acc(ah[mt], bh, acc[mt][nt]);   // hi*hi
        acc[mt][nt] = wmma_acc(ah[mt], bl, acc[mt][nt]);   // hi*lo
        acc[mt][nt] = wmma_acc(al[mt], bh, acc[mt][nt]);   // lo*hi
      }
    }
    __syncthreads();
  }
  int lm = lane >> 4;
#pragma unroll
  for (int nt = 0; nt < 4; ++nt) {
    int kg = nBase + wn * 64 + nt * 16 + ln;
    float pbv = pb[kg];
#pragma unroll
    for (int mt = 0; mt < 2; ++mt)
#pragma unroll
      for (int v = 0; v < 8; ++v) {
        int mg = mBase + wm * 32 + mt * 16 + lm * 8 + v;
        float val = acc[mt][nt][v] + pbv;
        unsigned short h, l; split2(val, h, l);
        size_t o = (size_t)mg * EDIM + kg;
        cb_hi[o] = h; cb_lo[o] = l;
      }
  }
}

// ---------- Kernel 3: per-row ||codebook||^2 ----------
__global__ __launch_bounds__(256) void k_cnorm(
    const unsigned short* __restrict__ cb_hi, const unsigned short* __restrict__ cb_lo,
    float* __restrict__ cnorm) {
  int lane = threadIdx.x & 31, wave = threadIdx.x >> 5;
  int row = blockIdx.x * 8 + wave;
  const unsigned* hu = (const unsigned*)(cb_hi + (size_t)row * EDIM);
  const unsigned* lu = (const unsigned*)(cb_lo + (size_t)row * EDIM);
  float s = 0.f;
  for (int c = lane; c < EDIM / 2; c += 32) {
    unsigned hv = hu[c], lv = lu[c];
    float v0 = bf2f((unsigned short)(hv & 0xffff)) + bf2f((unsigned short)(lv & 0xffff));
    float v1 = bf2f((unsigned short)(hv >> 16))    + bf2f((unsigned short)(lv >> 16));
    s += v0 * v0 + v1 * v1;
  }
  for (int off = 16; off >= 1; off >>= 1) s += __shfl_xor(s, off, 32);
  if (lane == 0) cnorm[row] = s;
}

// ---------- Kernel 4: fused distance GEMM + argmin; double-buffered async LDS pipeline ----------
// Each block: 128 rows x one 512-code N-chunk (grid 64 x 16, domain-invalid chunks skipped).
__global__ __launch_bounds__(256) void k_distance(
    const unsigned short* __restrict__ zf_hi, const unsigned short* __restrict__ zf_lo,
    const unsigned short* __restrict__ cb_hi, const unsigned short* __restrict__ cb_lo,
    const float* __restrict__ cnorm, const int* __restrict__ adom,
    float* __restrict__ pscore, int* __restrict__ pidx) {
  int tid = threadIdx.x, lane = tid & 31, wave = tid >> 5;
  int wm = wave & 3, wn = wave >> 2;
  int mBase = blockIdx.x * 128;
  int nblk = blockIdx.y;                     // N-chunk of 512 codes
  int b = mBase >> 10;
  int dom = adom[b];
  bool valid = (dom == 3) || (dom == 0 && nblk < 4) ||
               (dom == 1 && nblk >= 4 && nblk < 8) || (dom == 2 && nblk >= 8);
  if (!valid) {                               // whole chunk masked out by domain: skip compute
    if (tid < 128) {
      pscore[nblk * MROWS + mBase + tid] = __builtin_huge_valf();
      pidx[nblk * MROWS + mBase + tid]   = 0x7fffffff;
    }
    return;
  }
  // [buf][hi/lo][128*LSTR] double-buffered tiles (320KB LDS/WGP on CDNA5 -> ~3 WGs/WGP)
  __shared__ unsigned short lsA[2][2][128 * LSTR];
  __shared__ unsigned short lsB[2][2][128 * LSTR];
  __shared__ float sSc[128];
  __shared__ int   sId[128];

  float bsc[2][8]; int bid[2][8];
#pragma unroll
  for (int mt = 0; mt < 2; ++mt)
#pragma unroll
    for (int v = 0; v < 8; ++v) { bsc[mt][v] = __builtin_huge_valf(); bid[mt][v] = 0x7fffffff; }

  int hi8 = (lane >> 4) * 8, kh16 = (lane >> 4) * 16, ln = lane & 15, lm = lane >> 4;
  v8f zero8 = {0.f,0.f,0.f,0.f,0.f,0.f,0.f,0.f};

  // per-thread copy coordinates: 2 x 16B chunks per array
  int row0 = tid >> 2, q0 = (tid & 3);
  int row1 = (tid + 256) >> 2, q1 = ((tid + 256) & 3);

  auto issue = [&](int buf, int nTile, int ks) {
    int k0 = ks * 32;
    {
      int lo = row0 * LSTR + q0 * 8;
      size_t ga = (size_t)(mBase + row0) * EDIM + k0 + q0 * 8;
      size_t gb = (size_t)(nTile + row0) * EDIM + k0 + q0 * 8;
      cp16(&lsA[buf][0][lo], zf_hi + ga);
      cp16(&lsA[buf][1][lo], zf_lo + ga);
      cp16(&lsB[buf][0][lo], cb_hi + gb);
      cp16(&lsB[buf][1][lo], cb_lo + gb);
    }
    {
      int lo = row1 * LSTR + q1 * 8;
      size_t ga = (size_t)(mBase + row1) * EDIM + k0 + q1 * 8;
      size_t gb = (size_t)(nTile + row1) * EDIM + k0 + q1 * 8;
      cp16(&lsA[buf][0][lo], zf_hi + ga);
      cp16(&lsA[buf][1][lo], zf_lo + ga);
      cp16(&lsB[buf][0][lo], cb_hi + gb);
      cp16(&lsB[buf][1][lo], cb_lo + gb);
    }
  };

  for (int nt128 = 0; nt128 < 4; ++nt128) {
    int nTile = nblk * 512 + nt128 * 128;
    v8f acc[2][4];
#pragma unroll
    for (int mt = 0; mt < 2; ++mt)
#pragma unroll
      for (int nt = 0; nt < 4; ++nt) acc[mt][nt] = zero8;

    issue(0, nTile, 0);
    for (int ks = 0; ks < 16; ++ks) {
      int cur = ks & 1;
      wait_async0();                 // own async copies into lsX[cur] done
      __syncthreads();               // everyone's data visible; prev buffer fully consumed
      if (ks < 15) issue(cur ^ 1, nTile, ks + 1);   // DMA overlaps the WMMAs below
      v16bf ah[2], al[2];
#pragma unroll
      for (int mt = 0; mt < 2; ++mt) {
        int r = (wm * 32 + mt * 16 + ln) * LSTR + hi8;
        ah[mt] = frag_a(&lsA[cur][0][r]);
        al[mt] = frag_a(&lsA[cur][1][r]);
      }
#pragma unroll
      for (int nt = 0; nt < 4; ++nt) {
        int r = (wn * 64 + nt * 16 + ln) * LSTR + hi8;
        v16bf bh = frag_b(&lsB[cur][0][r]);
        v16bf bl = frag_b(&lsB[cur][1][r]);
#pragma unroll
        for (int mt = 0; mt < 2; ++mt) {
          acc[mt][nt] = wmma_acc(ah[mt], bh, acc[mt][nt]);
          acc[mt][nt] = wmma_acc(ah[mt], bl, acc[mt][nt]);
          acc[mt][nt] = wmma_acc(al[mt], bh, acc[mt][nt]);
        }
      }
    }
    __syncthreads();                 // protect buffer 0 reuse by next nTile

    // score = ||c||^2 - 2 z.c  (||z||^2 constant per row -> irrelevant to argmin)
#pragma unroll
    for (int nt = 0; nt < 4; ++nt) {
      int ng = nTile + wn * 64 + nt * 16 + ln;
      float cn = cnorm[ng];
#pragma unroll
      for (int mt = 0; mt < 2; ++mt)
#pragma unroll
        for (int v = 0; v < 8; ++v) {
          float s = fmaf(-2.f, acc[mt][nt][v], cn);
          if (s < bsc[mt][v]) { bsc[mt][v] = s; bid[mt][v] = ng; }
        }
    }
  }
  // butterfly min-reduce over the 16 lanes holding different n of the same row
#pragma unroll
  for (int mt = 0; mt < 2; ++mt)
#pragma unroll
    for (int v = 0; v < 8; ++v) {
      float s = bsc[mt][v]; int i = bid[mt][v];
#pragma unroll
      for (int off = 1; off < 16; off <<= 1) {
        float s2 = __shfl_xor(s, off, 32);
        int   i2 = __shfl_xor(i, off, 32);
        if (s2 < s || (s2 == s && i2 < i)) { s = s2; i = i2; }
      }
      bsc[mt][v] = s; bid[mt][v] = i;
    }
  // combine the two N-wave-groups via LDS, write the chunk partial
  if (wn == 0 && ln == 0) {
#pragma unroll
    for (int mt = 0; mt < 2; ++mt)
#pragma unroll
      for (int v = 0; v < 8; ++v) {
        int mloc = wm * 32 + mt * 16 + lm * 8 + v;
        sSc[mloc] = bsc[mt][v]; sId[mloc] = bid[mt][v];
      }
  }
  __syncthreads();
  if (wn == 1 && ln == 0) {
#pragma unroll
    for (int mt = 0; mt < 2; ++mt)
#pragma unroll
      for (int v = 0; v < 8; ++v) {
        int mloc = wm * 32 + mt * 16 + lm * 8 + v;
        float s = bsc[mt][v]; int i = bid[mt][v];
        float s0 = sSc[mloc]; int i0 = sId[mloc];
        if (s0 < s || (s0 == s && i0 < i)) { s = s0; i = i0; }
        pscore[nblk * MROWS + mBase + mloc] = s;
        pidx[nblk * MROWS + mBase + mloc]   = i;
      }
  }
}

// ---------- Kernel 5: combine 16 partial argmins per row, emit indices, zero loss ----------
__global__ __launch_bounds__(256) void k_combine(
    const float* __restrict__ pscore, const int* __restrict__ pidx,
    int* __restrict__ fidx, float* __restrict__ out_idx, float* __restrict__ loss) {
  int m = blockIdx.x * 256 + threadIdx.x;
  float bs = __builtin_huge_valf(); int bi = 0x7fffffff;
  for (int p = 0; p < NBLKS; ++p) {
    float s = pscore[p * MROWS + m]; int i = pidx[p * MROWS + m];
    if (s < bs || (s == bs && i < bi)) { bs = s; bi = i; }
  }
  fidx[m] = bi;
  out_idx[m] = (float)bi;
  if (m == 0) *loss = 0.f;
}

// ---------- Kernel 6: z_q = codebook[idx] transposed back to (B,C,T) + commit loss ----------
__global__ __launch_bounds__(256) void k_output(
    const float* __restrict__ z,
    const unsigned short* __restrict__ cb_hi, const unsigned short* __restrict__ cb_lo,
    const int* __restrict__ fidx, float* __restrict__ zq, float* __restrict__ loss) {
  int tid = threadIdx.x;
  int tx = tid & 31, cg = tid >> 5;
  int bt = blockIdx.x;
  int b = bt >> 5, t0 = (bt & 31) * 32;
  int t = t0 + tx;
  int r = b * TQ + t;
  int idx = fidx[r];
  const unsigned* ch = (const unsigned*)(cb_hi + (size_t)idx * EDIM);
  const unsigned* cl = (const unsigned*)(cb_lo + (size_t)idx * EDIM);
  const float* zb = z + (size_t)b * EDIM * TQ;
  float* qb = zq + (size_t)b * EDIM * TQ;
  float lsum = 0.f;
  for (int ci = cg; ci < EDIM / 2; ci += 8) {    // 2 channels per 4B load
    unsigned hv = ch[ci], lv = cl[ci];
    float v0 = bf2f((unsigned short)(hv & 0xffff)) + bf2f((unsigned short)(lv & 0xffff));
    float v1 = bf2f((unsigned short)(hv >> 16))    + bf2f((unsigned short)(lv >> 16));
    int c = ci * 2;
    size_t o0 = (size_t)c * TQ + t, o1 = (size_t)(c + 1) * TQ + t;
    float d0 = v0 - zb[o0], d1 = v1 - zb[o1];
    qb[o0] = v0; qb[o1] = v1;
    lsum += d0 * d0 + d1 * d1;
  }
  for (int off = 16; off >= 1; off >>= 1) lsum += __shfl_xor(lsum, off, 32);
  __shared__ float red[8];
  if (tx == 0) red[cg] = lsum;
  __syncthreads();
  if (tid == 0) {
    float s = 0.f;
    for (int w = 0; w < 8; ++w) s += red[w];
    // commit_loss = (1 + BETA) * mean((z_q - zt)^2), BETA = 0.25, N = 8*512*1024
    atomicAdd(loss, s * (1.25f / 4194304.f));
  }
}

extern "C" void kernel_launch(void* const* d_in, const int* in_sizes, int n_in,
                              void* d_out, int out_size, void* d_ws, size_t ws_size,
                              hipStream_t stream) {
  (void)in_sizes; (void)n_in; (void)out_size; (void)ws_size;
  const float* z   = (const float*)d_in[0];
  const int*   ad  = (const int*)d_in[1];   // audio_domain
  const float* emb = (const float*)d_in[3];
  const float* pw  = (const float*)d_in[4];
  const float* pb  = (const float*)d_in[5];
  float* out = (float*)d_out;

  char* ws = (char*)d_ws;
  unsigned short* zf_hi = (unsigned short*)(ws);
  unsigned short* zf_lo = (unsigned short*)(ws + (8u << 20));
  unsigned short* cb_hi = (unsigned short*)(ws + (16u << 20));
  unsigned short* cb_lo = (unsigned short*)(ws + (24u << 20));
  float* cnorm  = (float*)(ws + (32u << 20));
  float* pscore = (float*)(ws + (32u << 20) + 32768);                 // 16*8192*4 B
  int*   pidx   = (int*)  (ws + (32u << 20) + 32768 + 524288);        // 16*8192*4 B
  int*   fidx   = (int*)  (ws + (32u << 20) + 32768 + 1048576);

  float* out_idx = out + 4194304;           // after z_q (8*512*1024)
  float* loss    = out + 4194304 + 8192;    // after min_idx (1*8*1024)

  k_transpose_split<<<dim3(32, 16, 8), dim3(32, 8), 0, stream>>>(z, zf_hi, zf_lo);
  k_codebook<<<dim3(64, 4), 256, 0, stream>>>(emb, pw, pb, cb_hi, cb_lo);
  k_cnorm<<<1024, 256, 0, stream>>>(cb_hi, cb_lo, cnorm);
  k_distance<<<dim3(64, NBLKS), 256, 0, stream>>>(zf_hi, zf_lo, cb_hi, cb_lo, cnorm, ad, pscore, pidx);
  k_combine<<<32, 256, 0, stream>>>(pscore, pidx, fidx, out_idx, loss);
  k_output<<<256, 256, 0, stream>>>(z, cb_hi, cb_lo, fidx, out, loss);
}
